// Multihead_Self_Attention_15702400434587
// MI455X (gfx1250) — matrix operand order repaired and checked by reading, not compiled, not run on hardware
//
#include <hip/hip_runtime.h>
#include <math.h>

// ---------------------------------------------------------------------------
// MI455X (gfx1250, wave32) multi-head self-attention with RoPE + causal mask.
// All matmuls on v_wmma_f32_16x16x32_f16 (f16 in, f32 accumulate).
// ---------------------------------------------------------------------------

typedef __attribute__((ext_vector_type(16))) _Float16 v16h;
typedef __attribute__((ext_vector_type(8)))  _Float16 v8h;
typedef __attribute__((ext_vector_type(8)))  float    v8f;

#define B_   2
#define S_   2048
#define D_   1024
#define H_   16
#define DK_  64
#define M_   (B_ * S_)      // 4096 rows (b*s)
#define THETA_ 10000.0f

__device__ __forceinline__ v16h cat8(v8h lo, v8h hi) {
  return __builtin_shufflevector(lo, hi, 0,1,2,3,4,5,6,7,8,9,10,11,12,13,14,15);
}

__device__ __forceinline__ v8f wmma_f16(v16h a, v16h b, v8f c) {
  // D = A(16x32) * B(32x16) + C, fp32 accumulate
  return __builtin_amdgcn_wmma_f32_16x16x32_f16(false, a, false, b, (short)0, c,
                                                false, false);
}

// reductions within each 16-lane half (rows of the C/D fragment live per-half)
__device__ __forceinline__ float half_red_max(float v) {
  v = fmaxf(v, __shfl_xor(v, 1, 32));
  v = fmaxf(v, __shfl_xor(v, 2, 32));
  v = fmaxf(v, __shfl_xor(v, 4, 32));
  v = fmaxf(v, __shfl_xor(v, 8, 32));
  return v;
}
__device__ __forceinline__ float half_red_sum(float v) {
  v += __shfl_xor(v, 1, 32);
  v += __shfl_xor(v, 2, 32);
  v += __shfl_xor(v, 4, 32);
  v += __shfl_xor(v, 8, 32);
  return v;
}

// ---------------------------------------------------------------------------
// GEMM:  Y[m, n] = sum_k A[m, k] * W[n, k]     (M=4096, N=K=1024)
// A is fp32 (converted to f16 on the fly) or f16.  8 waves/block, each wave
// owns a 16x64 output strip; K-loop of 32 feeding 4 WMMAs per step.
// OUT_MODE 0: fp32 Y[m*D + n]
// OUT_MODE 1: f16 transposed per-head V:  Vt[((b*H + h)*DK + d)*S + s]
// ---------------------------------------------------------------------------
template <bool A_F16, int OUT_MODE>
__global__ __launch_bounds__(256) void gemm_xWT(const void* __restrict__ Aptr,
                                                const float* __restrict__ W,
                                                void* __restrict__ Out) {
  const int lane = threadIdx.x & 31;
  const int wave = threadIdx.x >> 5;
  const int m0   = blockIdx.y * 128 + wave * 16;
  const int n0   = blockIdx.x * 64;
  const int nlo  = lane & 15;
  const int hih  = lane >> 4;          // which 16-lane half
  const int kb0a = hih ? 8 : 0;        // A-fragment K swizzle
  const int kb0b = hih ? 8 : 0;        // B-fragment K swizzle
  const size_t arow = (size_t)(m0 + nlo) * D_;

  v8f acc[4] = {};

  for (int k = 0; k < D_; k += 32) {
    v16h a;
    if (A_F16) {
      const _Float16* A16 = (const _Float16*)Aptr;
      v8h lo = *(const v8h*)(A16 + arow + k + kb0a);
      v8h hi = *(const v8h*)(A16 + arow + k + kb0a + 16);
      a = cat8(lo, hi);
    } else {
      const float* A32 = (const float*)Aptr;
      v8f lo = *(const v8f*)(A32 + arow + k + kb0a);
      v8f hi = *(const v8f*)(A32 + arow + k + kb0a + 16);
      a = cat8(__builtin_convertvector(lo, v8h), __builtin_convertvector(hi, v8h));
    }
#pragma unroll
    for (int t = 0; t < 4; ++t) {
      const float* wp = W + (size_t)(n0 + t * 16 + nlo) * D_ + k + kb0b;
      v8f w0 = *(const v8f*)(wp);
      v8f w1 = *(const v8f*)(wp + 16);
      v16h b = cat8(__builtin_convertvector(w0, v8h), __builtin_convertvector(w1, v8h));
      acc[t] = wmma_f16(a, b, acc[t]);
    }
  }

#pragma unroll
  for (int t = 0; t < 4; ++t) {
    const int n = n0 + t * 16 + nlo;
#pragma unroll
    for (int j = 0; j < 8; ++j) {
      const int m = m0 + j + 8 * hih;
      const float v = acc[t][j];
      if (OUT_MODE == 0) {
        ((float*)Out)[(size_t)m * D_ + n] = v;
      } else {
        const int bi = m / S_, s = m % S_;
        const int h = n >> 6, d = n & 63;
        ((_Float16*)Out)[(((size_t)bi * H_ + h) * DK_ + d) * S_ + s] = (_Float16)v;
      }
    }
  }
}

// ---------------------------------------------------------------------------
// RoPE + cast + head transpose:  Q32/K32 [m, e] fp32  ->  Qh/Kh [b,h,s,d] f16
// ---------------------------------------------------------------------------
__global__ void rope_cast(const float* __restrict__ Q32,
                          const float* __restrict__ K32,
                          const int* __restrict__ pos,
                          _Float16* __restrict__ Qh, _Float16* __restrict__ Kh) {
  const int idx = blockIdx.x * blockDim.x + threadIdx.x;  // one thread per pair
  if (idx >= M_ * (D_ / 2)) return;
  const int row = idx / (D_ / 2);
  const int pr  = idx % (D_ / 2);
  const int h   = pr / (DK_ / 2);
  const int i   = pr % (DK_ / 2);
  const int bi = row / S_, s = row % S_;

  const float p = (float)pos[s];
  const float inv_freq = __expf(-(2.0f * (float)i / (float)DK_) * __logf(THETA_));
  const float fr = p * inv_freq;
  const float c = __cosf(fr), sn = __sinf(fr);

  const size_t in0 = (size_t)row * D_ + h * DK_ + 2 * i;
  const size_t out0 = (((size_t)bi * H_ + h) * S_ + s) * DK_ + 2 * i;

  float x1 = Q32[in0], x2 = Q32[in0 + 1];
  Qh[out0]     = (_Float16)(x1 * c - x2 * sn);
  Qh[out0 + 1] = (_Float16)(x1 * sn + x2 * c);

  x1 = K32[in0]; x2 = K32[in0 + 1];
  Kh[out0]     = (_Float16)(x1 * c - x2 * sn);
  Kh[out0 + 1] = (_Float16)(x1 * sn + x2 * c);
}

// ---------------------------------------------------------------------------
// Causal flash attention.  One wave per 16-query tile; key blocks of 32.
// Q·K^T: two 16x16 score tiles via 2 K=32 WMMAs each.  Online softmax in f32,
// P staged through a wave-private LDS tile to form the A-fragment, then P·V
// with V stored transposed (key-contiguous) for clean B-fragment loads.
// ---------------------------------------------------------------------------
__global__ __launch_bounds__(256) void flash_attn(const _Float16* __restrict__ Qh,
                                                  const _Float16* __restrict__ Kh,
                                                  const _Float16* __restrict__ Vt,
                                                  _Float16* __restrict__ Oh) {
  __shared__ _Float16 Pl[8][16][32];  // per-wave P tile (16 queries x 32 keys)

  const int lane = threadIdx.x & 31;
  const int wave = threadIdx.x >> 5;
  const int h  = blockIdx.y;
  const int bi = blockIdx.z;
  const int q0 = (blockIdx.x * 8 + wave) * 16;
  const int nlo = lane & 15;
  const int hih = lane >> 4;
  const int kb0a = hih ? 8 : 0;
  const int kb0b = hih ? 8 : 0;

  const _Float16* Qb = Qh + (((size_t)bi * H_ + h) * S_) * DK_;
  const _Float16* Kb = Kh + (((size_t)bi * H_ + h) * S_) * DK_;
  const _Float16* Vb = Vt + (((size_t)bi * H_ + h) * DK_) * S_;

  // Q A-fragments for both 32-wide halves of dk=64
  v16h aq[2];
#pragma unroll
  for (int dh = 0; dh < 2; ++dh) {
    const _Float16* qp = Qb + (size_t)(q0 + nlo) * DK_ + dh * 32 + kb0a;
    aq[dh] = cat8(*(const v8h*)qp, *(const v8h*)(qp + 16));
  }

  v8f o[4] = {};
  float mrun[8], lrun[8];
#pragma unroll
  for (int j = 0; j < 8; ++j) { mrun[j] = -INFINITY; lrun[j] = 0.0f; }

  const int kend = q0 + 16;  // exclusive key bound (causal)
  for (int kb = 0; kb < kend; kb += 32) {
    // ---- scores: S = Q * K^T * 1/sqrt(dk), two 16x16 tiles --------------
    v8f sc[2] = {};
#pragma unroll
    for (int sub = 0; sub < 2; ++sub) {
#pragma unroll
      for (int dh = 0; dh < 2; ++dh) {
        const _Float16* kp =
            Kb + (size_t)(kb + sub * 16 + nlo) * DK_ + dh * 32 + kb0b;
        v16h bf = cat8(*(const v8h*)kp, *(const v8h*)(kp + 16));
        sc[sub] = wmma_f16(aq[dh], bf, sc[sub]);
      }
    }

    // ---- online softmax update ------------------------------------------
#pragma unroll
    for (int j = 0; j < 8; ++j) {
      const int q = q0 + j + 8 * hih;
      float v0 = sc[0][j] * 0.125f;   // 1/sqrt(64)
      float v1 = sc[1][j] * 0.125f;
      if (kb + nlo > q)      v0 = -INFINITY;   // causal mask
      if (kb + 16 + nlo > q) v1 = -INFINITY;
      const float mx = half_red_max(fmaxf(v0, v1));
      const float mnew = fmaxf(mrun[j], mx);
      const float f = __expf(mrun[j] - mnew);
      const float p0 = __expf(v0 - mnew);
      const float p1 = __expf(v1 - mnew);
      lrun[j] = lrun[j] * f + half_red_sum(p0 + p1);
      mrun[j] = mnew;
#pragma unroll
      for (int t = 0; t < 4; ++t) o[t][j] *= f;
      Pl[wave][j + 8 * hih][nlo]      = (_Float16)p0;
      Pl[wave][j + 8 * hih][16 + nlo] = (_Float16)p1;
    }

    // ---- O += P * V  (K=32 keys, N=64 dims as 4 tiles) ------------------
    const _Float16* prow = &Pl[wave][nlo][0];
    v16h pa = cat8(*(const v8h*)(prow + kb0a), *(const v8h*)(prow + kb0a + 16));
#pragma unroll
    for (int t = 0; t < 4; ++t) {
      const _Float16* vp = Vb + (size_t)(t * 16 + nlo) * S_ + kb + kb0b;
      v16h bv = cat8(*(const v8h*)vp, *(const v8h*)(vp + 16));
      o[t] = wmma_f16(pa, bv, o[t]);
    }
  }

  // ---- normalize and emit O in [b, s, e] layout (f16 for final GEMM) ----
#pragma unroll
  for (int j = 0; j < 8; ++j) {
    const int q = q0 + j + 8 * hih;
    const float inv = 1.0f / lrun[j];
    const size_t rowoff = ((size_t)bi * S_ + q) * D_ + h * DK_;
#pragma unroll
    for (int t = 0; t < 4; ++t)
      Oh[rowoff + t * 16 + nlo] = (_Float16)(o[t][j] * inv);
  }
}

// ---------------------------------------------------------------------------
extern "C" void kernel_launch(void* const* d_in, const int* in_sizes, int n_in,
                              void* d_out, int out_size, void* d_ws, size_t ws_size,
                              hipStream_t stream) {
  const float* x   = (const float*)d_in[0];
  const float* Wq  = (const float*)d_in[1];
  const float* Wk  = (const float*)d_in[2];
  const float* Wv  = (const float*)d_in[3];
  const float* Wo  = (const float*)d_in[4];
  const int*   pos = (const int*)d_in[5];

  // workspace carve-up (64 MB total)
  char* ws = (char*)d_ws;
  float*    Q32 = (float*)(ws);                           // 16 MB fp32
  float*    K32 = (float*)(ws + ((size_t)16 << 20));      // 16 MB fp32
  _Float16* Qh  = (_Float16*)(ws + ((size_t)32 << 20));   // 8 MB f16 [b,h,s,d]
  _Float16* Kh  = (_Float16*)(ws + ((size_t)40 << 20));   // 8 MB f16 [b,h,s,d]
  _Float16* Vt  = (_Float16*)(ws + ((size_t)48 << 20));   // 8 MB f16 [b,h,d,s]
  _Float16* Oh  = (_Float16*)(ws + ((size_t)56 << 20));   // 8 MB f16 [b,s,e]
  float* out = (float*)d_out;

  const dim3 gg(D_ / 64, M_ / 128);  // (16, 32) blocks of 8 waves
  gemm_xWT<false, 0><<<gg, 256, 0, stream>>>((const void*)x, Wq, (void*)Q32);
  gemm_xWT<false, 0><<<gg, 256, 0, stream>>>((const void*)x, Wk, (void*)K32);
  gemm_xWT<false, 1><<<gg, 256, 0, stream>>>((const void*)x, Wv, (void*)Vt);

  const int pairs = M_ * (D_ / 2);
  rope_cast<<<(pairs + 255) / 256, 256, 0, stream>>>(Q32, K32, pos, Qh, Kh);

  const dim3 ga(S_ / 16 / 8, H_, B_);  // (16, 16, 2)
  flash_attn<<<ga, 256, 0, stream>>>(Qh, Kh, Vt, Oh);

  gemm_xWT<true, 0><<<gg, 256, 0, stream>>>((const void*)Oh, Wo, (void*)out);
}
